// QwenMoeWrapperSkipAttn_32461362823837
// MI455X (gfx1250) — matrix operand order repaired and checked by reading, not compiled, not run on hardware
//
#include <hip/hip_runtime.h>
#include <hip/hip_bf16.h>

// ---------------- problem constants ----------------
#define T_TOK 4096   // B*S
#define DIM   1024   // D
#define NEXP  16     // E
#define FDIM  512    // F

// ---------------- WMMA types ----------------
typedef __attribute__((ext_vector_type(16))) __bf16 v16bf;
typedef __attribute__((ext_vector_type(8)))  float  v8f;
typedef __attribute__((ext_vector_type(4)))  unsigned int v4u;
typedef __attribute__((ext_vector_type(4)))  int v4i;

union ABFrag { v16bf bf; v4u u[2]; };

static __device__ __forceinline__ v8f wmma_bf16(const ABFrag& a, const ABFrag& b, v8f c) {
  return __builtin_amdgcn_wmma_f32_16x16x32_bf16(false, a.bf, false, b.bf,
                                                 (short)0, c, false, false);
}

// fp32 -> bf16 bits (round to nearest even)
static __device__ __forceinline__ unsigned short f2bf(float f) {
  unsigned u = __float_as_uint(f);
  unsigned r = u + 0x7fffu + ((u >> 16) & 1u);
  return (unsigned short)(r >> 16);
}

// ---------------- async global->LDS copy (CDNA5), guarded fallback ----------------
#define USE_ASYNC_LDS 0
#if defined(__has_builtin)
#if __has_builtin(__builtin_amdgcn_global_load_async_to_lds_b128) && \
    __has_builtin(__builtin_amdgcn_s_wait_asynccnt)
#undef USE_ASYNC_LDS
#define USE_ASYNC_LDS 1
#endif
#endif

static __device__ __forceinline__ void cp_b128(unsigned short* lds_dst,
                                               const unsigned short* gsrc) {
#if USE_ASYNC_LDS
  __builtin_amdgcn_global_load_async_to_lds_b128(
      (__attribute__((address_space(1))) v4i*)(void*)gsrc,
      (__attribute__((address_space(3))) v4i*)(void*)lds_dst, 0, 0);
#else
  *(v4u*)lds_dst = *(const v4u*)gsrc;
#endif
}

static __device__ __forceinline__ void cp_wait() {
#if USE_ASYNC_LDS
  __builtin_amdgcn_s_wait_asynccnt(0);
#endif
}

// ---------------- small kernels ----------------
__global__ void init_counts_kernel(int* counts, int* fill) {
  int t = threadIdx.x;
  if (t < NEXP) { counts[t] = 0; fill[t] = 0; }
}

__global__ __launch_bounds__(256) void cvt_x_kernel(const float* __restrict__ x,
                                                    unsigned short* __restrict__ xbf) {
  int idx = (blockIdx.x * 256 + threadIdx.x) * 4;   // T*D divisible by 4
  float4 v = *(const float4*)(x + idx);
  unsigned long long p = (unsigned long long)f2bf(v.x)
                       | ((unsigned long long)f2bf(v.y) << 16)
                       | ((unsigned long long)f2bf(v.z) << 32)
                       | ((unsigned long long)f2bf(v.w) << 48);
  *(unsigned long long*)(xbf + idx) = p;
}

// Transpose + convert: in (E stacked) M x N fp32 row-major -> out N x M bf16 row-major.
// grid: (N/32, M/32, E), 256 threads.
__global__ __launch_bounds__(256) void transpose_cvt_kernel(const float* __restrict__ in,
                                                            unsigned short* __restrict__ out,
                                                            int M, int N) {
  __shared__ float tile[32][33];
  const int e = blockIdx.z;
  const int n0 = blockIdx.x * 32;
  const int m0 = blockIdx.y * 32;
  const int tx = threadIdx.x & 31;
  const int ty = threadIdx.x >> 5;          // 0..7
  const float* src = in + (size_t)e * M * N;
  unsigned short* dst = out + (size_t)e * N * M;
#pragma unroll
  for (int i = 0; i < 4; ++i)
    tile[ty + i * 8][tx] = src[(size_t)(m0 + ty + i * 8) * N + n0 + tx];
  __syncthreads();
#pragma unroll
  for (int i = 0; i < 4; ++i)
    dst[(size_t)(n0 + ty + i * 8) * M + m0 + tx] = f2bf(tile[tx][ty + i * 8]);
}

__global__ __launch_bounds__(256) void zero_out_kernel(float* __restrict__ out) {
  int idx = (blockIdx.x * 256 + threadIdx.x) * 4;
  float4 z = {0.f, 0.f, 0.f, 0.f};
  *(float4*)(out + idx) = z;
}

// ---------------- router ----------------
__global__ __launch_bounds__(128) void router_kernel(const float* __restrict__ x,
                                                     const float* __restrict__ gate_w,
                                                     const int* __restrict__ tokmod,
                                                     const int* __restrict__ expmod,
                                                     int* __restrict__ top_idx,
                                                     float* __restrict__ top_w,
                                                     int* __restrict__ counts) {
  __shared__ float xs[DIM];
  __shared__ float lg[NEXP];
  const int t = blockIdx.x;
  const int tid = threadIdx.x;
  const float* xr = x + (size_t)t * DIM;
  for (int i = tid; i < DIM; i += 128) xs[i] = xr[i];
  __syncthreads();

  const int e = tid >> 3, part = tid & 7;
  const float* gw = gate_w + (size_t)e * DIM;
  float s = 0.f;
  const int k0 = part * 128;
  for (int k = k0; k < k0 + 128; ++k) s += xs[k] * gw[k];
  s += __shfl_xor(s, 1);
  s += __shfl_xor(s, 2);
  s += __shfl_xor(s, 4);
  if (part == 0) lg[e] = s;
  __syncthreads();

  if (tid == 0) {
    float mx = lg[0];
    for (int i = 1; i < NEXP; ++i) mx = fmaxf(mx, lg[i]);
    float p[NEXP];
    for (int i = 0; i < NEXP; ++i) p[i] = __expf(lg[i] - mx);
    int i1 = 0; float b1 = p[0];
    for (int i = 1; i < NEXP; ++i) if (p[i] > b1) { b1 = p[i]; i1 = i; }
    int i2 = (i1 == 0) ? 1 : 0; float b2 = p[i2];
    for (int i = 0; i < NEXP; ++i) if (i != i1 && p[i] > b2) { b2 = p[i]; i2 = i; }
    float sum = b1 + b2;
    float w1 = b1 / sum, w2 = b2 / sum;
    const int tm = tokmod[t];
    const int m1 = expmod[i1], m2 = expmod[i2];
    if (tm != 0 && m1 != 0 && tm * m1 == -1) w1 = 0.f;
    if (tm != 0 && m2 != 0 && tm * m2 == -1) w2 = 0.f;
    const float s2 = w1 + w2;
    if (s2 > 0.f) { float d = fmaxf(s2, 1e-9f); w1 /= d; w2 /= d; }
    top_idx[t * 2 + 0] = i1; top_idx[t * 2 + 1] = i2;
    top_w[t * 2 + 0] = w1;  top_w[t * 2 + 1] = w2;
    atomicAdd(&counts[i1], 1);
    atomicAdd(&counts[i2], 1);
  }
}

__global__ void scan_kernel(const int* __restrict__ counts, int* __restrict__ offsets) {
  if (threadIdx.x == 0) {
    int acc = 0;
    for (int i = 0; i < NEXP; ++i) { offsets[i] = acc; acc += counts[i]; }
    offsets[NEXP] = acc;
  }
}

__global__ __launch_bounds__(256) void scatter_kernel(const int* __restrict__ top_idx,
                                                      const float* __restrict__ top_w,
                                                      const int* __restrict__ offsets,
                                                      int* __restrict__ fill,
                                                      int* __restrict__ atok,
                                                      float* __restrict__ aw) {
  int t = blockIdx.x * 256 + threadIdx.x;
  if (t >= T_TOK) return;
  for (int k = 0; k < 2; ++k) {
    int e = top_idx[t * 2 + k];
    int pos = offsets[e] + atomicAdd(&fill[e], 1);
    atok[pos] = t;
    aw[pos] = top_w[t * 2 + k];
  }
}

// ---------------- GEMM cores ----------------
// LDS row pitch for K-tile of 64 bf16: 64+8 = 72 elems (144B, 16B aligned)
#define PITCH 72

// GEMM1: h = silu(X*Wg)*(X*Wu). grid (64 row-tiles, E, F/64). TM=64, TK=64.
__global__ __launch_bounds__(256) void moe_gemm1(const unsigned short* __restrict__ xbf,
                                                 const unsigned short* __restrict__ gupT, // E x 2F x D bf16
                                                 const int* __restrict__ offsets,
                                                 const int* __restrict__ atok,
                                                 unsigned short* __restrict__ hbuf) {
  const int e = blockIdx.y;
  const int beg = offsets[e];
  const int cnt = offsets[e + 1] - beg;
  const int rt = blockIdx.x * 64;
  if (rt >= cnt) return;                 // uniform exit: EXEC stays full for WMMA
  const int fz = blockIdx.z * 64;

  __shared__ unsigned short As[64 * PITCH];
  __shared__ unsigned short Bg[64 * PITCH];
  __shared__ unsigned short Bu[64 * PITCH];

  const int tid = threadIdx.x;
  const int lane = tid & 31;
  const int wave = tid >> 5;

  // tile loader: thread owns one row (tid/4) x one 32B chunk ((tid&3)*16 elems)
  const int trow = tid >> 2;            // 0..63
  const int tcol = (tid & 3) * 16;      // 0,16,32,48 (bf16 elems)
  int aloc = rt + trow; if (aloc >= cnt) aloc = cnt - 1;
  const unsigned short* aptr = xbf + (size_t)atok[beg + aloc] * DIM;
  const unsigned short* gptr = gupT + ((size_t)e * 2 * FDIM + fz + trow) * DIM;
  const unsigned short* uptr = gupT + ((size_t)e * 2 * FDIM + FDIM + fz + trow) * DIM;

  const int mt   = (wave >> 1) * 16;
  const int n32  = (wave & 1) * 32;
  const int half = lane >> 4;
  const int l16  = lane & 15;

  v8f accG[2] = {v8f{}, v8f{}};
  v8f accU[2] = {v8f{}, v8f{}};

  for (int k0 = 0; k0 < DIM; k0 += 64) {
    cp_b128(&As[trow * PITCH + tcol],     aptr + k0 + tcol);
    cp_b128(&As[trow * PITCH + tcol + 8], aptr + k0 + tcol + 8);
    cp_b128(&Bg[trow * PITCH + tcol],     gptr + k0 + tcol);
    cp_b128(&Bg[trow * PITCH + tcol + 8], gptr + k0 + tcol + 8);
    cp_b128(&Bu[trow * PITCH + tcol],     uptr + k0 + tcol);
    cp_b128(&Bu[trow * PITCH + tcol + 8], uptr + k0 + tcol + 8);
    cp_wait();
    __syncthreads();

#pragma unroll
    for (int ks = 0; ks < 64; ks += 32) {
      ABFrag a;   // A 16x32: VGPR0-3 K=ks+half*8.., VGPR4-7 K=ks+16+half*8..
      a.u[0] = *(const v4u*)(&As[(mt + l16) * PITCH + ks + half * 8]);
      a.u[1] = *(const v4u*)(&As[(mt + l16) * PITCH + ks + 16 + half * 8]);
#pragma unroll
      for (int j = 0; j < 2; ++j) {
        const int n = n32 + j * 16 + l16;
        ABFrag bg, bu;  // B 32x16: VGPR0-3 K=ks+half*8.., VGPR4-7 K=ks+16+half*8..
        bg.u[0] = *(const v4u*)(&Bg[n * PITCH + ks + half * 8]);
        bg.u[1] = *(const v4u*)(&Bg[n * PITCH + ks + 16 + half * 8]);
        bu.u[0] = *(const v4u*)(&Bu[n * PITCH + ks + half * 8]);
        bu.u[1] = *(const v4u*)(&Bu[n * PITCH + ks + 16 + half * 8]);
        accG[j] = wmma_bf16(a, bg, accG[j]);
        accU[j] = wmma_bf16(a, bu, accU[j]);
      }
    }
    __syncthreads();
  }

#pragma unroll
  for (int r = 0; r < 8; ++r) {
    const int m = mt + half * 8 + r;
    const int loc = rt + m;
    if (loc < cnt) {
#pragma unroll
      for (int j = 0; j < 2; ++j) {
        float g = accG[j][r];
        float u = accU[j][r];
        float h = g * u / (1.f + __expf(-g));
        int f = fz + n32 + j * 16 + l16;
        hbuf[(size_t)(beg + loc) * FDIM + f] = f2bf(h);
      }
    }
  }
}

// GEMM2: out[tok] += w * (h * down[e]). grid (64 row-tiles, E, D/64). K=512, TK=64.
__global__ __launch_bounds__(256) void moe_gemm2(const unsigned short* __restrict__ hbuf,
                                                 const unsigned short* __restrict__ downT, // E x D x F bf16
                                                 const int* __restrict__ offsets,
                                                 const int* __restrict__ atok,
                                                 const float* __restrict__ aw,
                                                 float* __restrict__ out) {
  const int e = blockIdx.y;
  const int beg = offsets[e];
  const int cnt = offsets[e + 1] - beg;
  const int rt = blockIdx.x * 64;
  if (rt >= cnt) return;
  const int dz = blockIdx.z * 64;

  __shared__ unsigned short As[64 * PITCH];
  __shared__ unsigned short Bs[64 * PITCH];

  const int tid = threadIdx.x;
  const int lane = tid & 31;
  const int wave = tid >> 5;

  const int trow = tid >> 2;
  const int tcol = (tid & 3) * 16;
  int aloc = rt + trow; if (aloc >= cnt) aloc = cnt - 1;
  const unsigned short* aptr = hbuf + (size_t)(beg + aloc) * FDIM;
  const unsigned short* bptr = downT + ((size_t)e * DIM + dz + trow) * FDIM;

  const int mt   = (wave >> 1) * 16;
  const int n32  = (wave & 1) * 32;
  const int half = lane >> 4;
  const int l16  = lane & 15;

  v8f acc[2] = {v8f{}, v8f{}};

  for (int k0 = 0; k0 < FDIM; k0 += 64) {
    cp_b128(&As[trow * PITCH + tcol],     aptr + k0 + tcol);
    cp_b128(&As[trow * PITCH + tcol + 8], aptr + k0 + tcol + 8);
    cp_b128(&Bs[trow * PITCH + tcol],     bptr + k0 + tcol);
    cp_b128(&Bs[trow * PITCH + tcol + 8], bptr + k0 + tcol + 8);
    cp_wait();
    __syncthreads();

#pragma unroll
    for (int ks = 0; ks < 64; ks += 32) {
      ABFrag a;
      a.u[0] = *(const v4u*)(&As[(mt + l16) * PITCH + ks + half * 8]);
      a.u[1] = *(const v4u*)(&As[(mt + l16) * PITCH + ks + 16 + half * 8]);
#pragma unroll
      for (int j = 0; j < 2; ++j) {
        const int n = n32 + j * 16 + l16;
        ABFrag b;
        b.u[0] = *(const v4u*)(&Bs[n * PITCH + ks + half * 8]);
        b.u[1] = *(const v4u*)(&Bs[n * PITCH + ks + 16 + half * 8]);
        acc[j] = wmma_bf16(a, b, acc[j]);
      }
    }
    __syncthreads();
  }

#pragma unroll
  for (int r = 0; r < 8; ++r) {
    const int m = mt + half * 8 + r;
    const int loc = rt + m;
    if (loc < cnt) {
      const int row = beg + loc;
      const float w = aw[row];
      const int tok = atok[row];
#pragma unroll
      for (int j = 0; j < 2; ++j) {
        int col = dz + n32 + j * 16 + l16;
        atomicAdd(&out[(size_t)tok * DIM + col], w * acc[j][r]);
      }
    }
  }
}

// ---------------- host launcher ----------------
extern "C" void kernel_launch(void* const* d_in, const int* in_sizes, int n_in,
                              void* d_out, int out_size, void* d_ws, size_t ws_size,
                              hipStream_t stream) {
  (void)in_sizes; (void)n_in; (void)out_size; (void)ws_size;
  const float* x        = (const float*)d_in[0];   // (B,S,D)
  const float* gate_w   = (const float*)d_in[1];   // (E,D)
  const float* gate_up  = (const float*)d_in[2];   // (E,D,2F)
  const float* down     = (const float*)d_in[3];   // (E,F,D)
  const int*   tokmod   = (const int*)d_in[4];     // (T,)
  const int*   expmod   = (const int*)d_in[5];     // (E,)
  float* out            = (float*)d_out;

  char* ws = (char*)d_ws;
  int*   counts  = (int*)  (ws + 0);
  int*   offsets = (int*)  (ws + 64);
  int*   fill    = (int*)  (ws + 192);
  int*   atok    = (int*)  (ws + 256);
  float* aw      = (float*)(ws + 33024);
  int*   top_idx = (int*)  (ws + 65792);
  float* top_w   = (float*)(ws + 98560);
  size_t off = 131328;
  unsigned short* xbf   = (unsigned short*)(ws + off); off += (size_t)T_TOK * DIM * 2;       // 8 MB
  unsigned short* hbuf  = (unsigned short*)(ws + off); off += (size_t)T_TOK * 2 * FDIM * 2;  // 8 MB
  unsigned short* gupT  = (unsigned short*)(ws + off); off += (size_t)NEXP * 2 * FDIM * DIM * 2; // 32 MB
  unsigned short* downT = (unsigned short*)(ws + off);                                       // 16 MB

  hipLaunchKernelGGL(init_counts_kernel, dim3(1), dim3(32), 0, stream, counts, fill);
  hipLaunchKernelGGL(cvt_x_kernel, dim3((T_TOK * DIM) / (256 * 4)), dim3(256), 0, stream, x, xbf);
  // gate_up: per-expert (D x 2F) -> (2F x D);  down: (F x D) -> (D x F)
  hipLaunchKernelGGL(transpose_cvt_kernel, dim3((2 * FDIM) / 32, DIM / 32, NEXP), dim3(256), 0,
                     stream, gate_up, gupT, DIM, 2 * FDIM);
  hipLaunchKernelGGL(transpose_cvt_kernel, dim3(DIM / 32, FDIM / 32, NEXP), dim3(256), 0,
                     stream, down, downT, FDIM, DIM);
  hipLaunchKernelGGL(router_kernel, dim3(T_TOK), dim3(128), 0, stream,
                     x, gate_w, tokmod, expmod, top_idx, top_w, counts);
  hipLaunchKernelGGL(scan_kernel, dim3(1), dim3(32), 0, stream, counts, offsets);
  hipLaunchKernelGGL(scatter_kernel, dim3(T_TOK / 256), dim3(256), 0, stream,
                     top_idx, top_w, offsets, fill, atok, aw);
  hipLaunchKernelGGL(zero_out_kernel, dim3((T_TOK * DIM) / (256 * 4)), dim3(256), 0, stream, out);
  hipLaunchKernelGGL(moe_gemm1, dim3(T_TOK / 64, NEXP, FDIM / 64), dim3(256), 0, stream,
                     xbf, gupT, offsets, atok, hbuf);
  hipLaunchKernelGGL(moe_gemm2, dim3(T_TOK / 64, NEXP, DIM / 64), dim3(256), 0, stream,
                     hbuf, downT, offsets, atok, aw, out);
}
